// QuantumKernelMethod__gen444_65481071404609
// MI455X (gfx1250) — matrix operand order repaired and checked by reading, not compiled, not run on hardware
//
#include <hip/hip_runtime.h>
#include <hip/hip_bf16.h>

// RBF kernel matrix: out[b,c] = exp(-max(0, xs[b] + ys[c] - 2 * sum_f gamma[f]*x[b,f]*y[c,f]))
// x: [4096,512] f32, y: [4096,512] f32, gamma: [512] f32, out: [4096,4096] f32.
// Cross term via bf16x3 split-precision WMMA (v_wmma_f32_16x16x32_bf16), double-buffered LDS.

typedef __attribute__((ext_vector_type(16))) __bf16 v16bf;
typedef __attribute__((ext_vector_type(8)))  float  v8f;

#define FDIM 512
#define NDIM 4096
#define BM 128
#define BN 128
#define BK 32
#define ROWU 20                  // uints per LDS row: 32 bf16 = 64B data, padded to 80B
#define BUFU (4 * BM * ROWU)     // uints per LDS buffer (AH|AL|BH|BL) = 40KB

union V16U { v16bf v; uint4 q[2]; };
union V8U  { v8f  v; float f[8]; };

__device__ __forceinline__ unsigned pack_bf16(float a, float b) {
  // truncating f32->bf16 pair (hi 16 bits); exact for the hi/lo split scheme
  return (__float_as_uint(a) >> 16) | (__float_as_uint(b) & 0xFFFF0000u);
}
__device__ __forceinline__ float trunc_bf16(float a) {
  return __uint_as_float(__float_as_uint(a) & 0xFFFF0000u);
}

// ---------------- prep: gamma-weighted row norms (one wave32 per row) ----------------
__global__ __launch_bounds__(256)
void qk_rownorm(const float* __restrict__ m, const float* __restrict__ gamma,
                float* __restrict__ out, int nrows) {
  const int lane = threadIdx.x & 31;
  const int wid  = threadIdx.x >> 5;
  const int row  = blockIdx.x * 8 + wid;
  if (row >= nrows) return;
  const float* r = m + (size_t)row * FDIM;
  float s = 0.0f;
  for (int f = lane; f < FDIM; f += 32) {
    float v = r[f];
    s = fmaf(gamma[f] * v, v, s);
  }
#pragma unroll
  for (int off = 16; off > 0; off >>= 1) s += __shfl_xor(s, off, 32);
  if (lane == 0) out[row] = s;
}

// Convert + store one K-tile (held in registers) into an LDS buffer.
__device__ __forceinline__ void stage_tile(unsigned* base, int rbase, int c4,
                                           const float4* px, const float4* py,
                                           float4 gv) {
#pragma unroll
  for (int i = 0; i < 4; ++i) {
    const int row = rbase + 32 * i;          // 0..127
    const float4 xv = px[i];
    const float4 yv = py[i];
    const float a0 = xv.x * gv.x, a1 = xv.y * gv.y, a2 = xv.z * gv.z, a3 = xv.w * gv.w;
    unsigned* p;
    p = base + row * ROWU;                               // A hi
    p[2 * c4]     = pack_bf16(a0, a1);
    p[2 * c4 + 1] = pack_bf16(a2, a3);
    p = base + BM * ROWU + row * ROWU;                   // A lo
    p[2 * c4]     = pack_bf16(a0 - trunc_bf16(a0), a1 - trunc_bf16(a1));
    p[2 * c4 + 1] = pack_bf16(a2 - trunc_bf16(a2), a3 - trunc_bf16(a3));
    p = base + 2 * BM * ROWU + row * ROWU;               // B hi
    p[2 * c4]     = pack_bf16(yv.x, yv.y);
    p[2 * c4 + 1] = pack_bf16(yv.z, yv.w);
    p = base + 3 * BM * ROWU + row * ROWU;               // B lo
    p[2 * c4]     = pack_bf16(yv.x - trunc_bf16(yv.x), yv.y - trunc_bf16(yv.y));
    p[2 * c4 + 1] = pack_bf16(yv.z - trunc_bf16(yv.z), yv.w - trunc_bf16(yv.w));
  }
}

// ---------------- main: 128x128 tile per block, bf16x3 WMMA GEMM + fused epilogue ----
__global__ __launch_bounds__(256)
void qk_gemm(const float* __restrict__ x, const float* __restrict__ y,
             const float* __restrict__ gamma, const float* __restrict__ xs,
             const float* __restrict__ ys, float* __restrict__ out) {
  __shared__ __align__(16) unsigned smem[2 * BUFU];      // 80 KB, double buffered

  const int t    = threadIdx.x;
  const int lane = t & 31;
  const int wid  = t >> 5;
  const int half = lane >> 4;                // 0: lanes 0-15, 1: lanes 16-31
  const int l16  = lane & 15;
  const int m0   = 32 * (wid & 3);           // wave's M origin within tile (4 waves down)
  const int n0   = 64 * (wid >> 2);          // wave's N origin within tile (2 waves across)

  const int blockM = blockIdx.y * BM;
  const int blockN = blockIdx.x * BN;

  const int c4    = t & 7;                   // which float4 of the 32-wide K row
  const int rbase = t >> 3;                  // 0..31

  const int NT = FDIM / BK;                  // 16 K-tiles

  v8f acc[2][4];
  {
    v8f z = {};
#pragma unroll
    for (int i = 0; i < 2; ++i)
#pragma unroll
      for (int j = 0; j < 4; ++j) acc[i][j] = z;
  }

  float4 px[4], py[4], pg;

  // Prologue: load + stage K-tile 0 into buffer 0.
  {
    const int f0 = 4 * c4;
    pg = *(const float4*)(gamma + f0);
#pragma unroll
    for (int i = 0; i < 4; ++i) {
      const int row = rbase + 32 * i;
      px[i] = *(const float4*)(x + (size_t)(blockM + row) * FDIM + f0);
      py[i] = *(const float4*)(y + (size_t)(blockN + row) * FDIM + f0);
    }
    stage_tile(smem, rbase, c4, px, py, pg);
  }
  __syncthreads();

  for (int kt = 0; kt < NT; ++kt) {
    const unsigned* base = smem + (kt & 1) * BUFU;
    const unsigned* AH = base;
    const unsigned* AL = base + 1 * BM * ROWU;
    const unsigned* BH = base + 2 * BM * ROWU;
    const unsigned* BL = base + 3 * BM * ROWU;

    // Issue next tile's global loads first; latency hides under the WMMAs below.
    if (kt + 1 < NT) {
      const int f0 = (kt + 1) * BK + 4 * c4;
      pg = *(const float4*)(gamma + f0);
#pragma unroll
      for (int i = 0; i < 4; ++i) {
        const int row = rbase + 32 * i;
        px[i] = *(const float4*)(x + (size_t)(blockM + row) * FDIM + f0);
        py[i] = *(const float4*)(y + (size_t)(blockN + row) * FDIM + f0);
      }
    }

    // A operands: ISA 16-bit A 16x32 layout -> bytes {0,32}+16*half of the M-row
    V16U ah[2], al[2];
#pragma unroll
    for (int mt = 0; mt < 2; ++mt) {
      const unsigned* rp  = AH + (m0 + 16 * mt + l16) * ROWU;
      const unsigned* rpl = AL + (m0 + 16 * mt + l16) * ROWU;
      ah[mt].q[0] = *(const uint4*)(rp + 4 * half);
      ah[mt].q[1] = *(const uint4*)(rp + 8 + 4 * half);
      al[mt].q[0] = *(const uint4*)(rpl + 4 * half);
      al[mt].q[1] = *(const uint4*)(rpl + 8 + 4 * half);
    }
#pragma unroll
    for (int nt = 0; nt < 4; ++nt) {
      // B operands: same layout as A -> bytes {0,32}+16*half of the N-row
      V16U bh, bl;
      const unsigned* rp  = BH + (n0 + 16 * nt + l16) * ROWU + 4 * half;
      const unsigned* rpl = BL + (n0 + 16 * nt + l16) * ROWU + 4 * half;
      bh.q[0] = *(const uint4*)(rp);
      bh.q[1] = *(const uint4*)(rp + 8);
      bl.q[0] = *(const uint4*)(rpl);
      bl.q[1] = *(const uint4*)(rpl + 8);
#pragma unroll
      for (int mt = 0; mt < 2; ++mt) {
        acc[mt][nt] = __builtin_amdgcn_wmma_f32_16x16x32_bf16(
            false, ah[mt].v, false, bh.v, (short)0, acc[mt][nt], false, false);  // hi*hi
        acc[mt][nt] = __builtin_amdgcn_wmma_f32_16x16x32_bf16(
            false, ah[mt].v, false, bl.v, (short)0, acc[mt][nt], false, false);  // hi*lo
        acc[mt][nt] = __builtin_amdgcn_wmma_f32_16x16x32_bf16(
            false, al[mt].v, false, bh.v, (short)0, acc[mt][nt], false, false);  // lo*hi
      }
    }

    // Stage the prefetched tile into the other buffer (safe: everyone passed the
    // previous barrier, so no wave is still reading it).
    if (kt + 1 < NT) {
      stage_tile(smem + ((kt + 1) & 1) * BUFU, rbase, c4, px, py, pg);
    }
    __syncthreads();
  }

  // Fused epilogue: C layout = lanes 0-15: M=i, lanes 16-31: M=8+i; N = lane%16
#pragma unroll
  for (int mt = 0; mt < 2; ++mt) {
#pragma unroll
    for (int nt = 0; nt < 4; ++nt) {
      V8U a; a.v = acc[mt][nt];
      const int col  = blockN + n0 + 16 * nt + l16;
      const float yc = ys[col];
      const int rowb = blockM + m0 + 16 * mt + 8 * half;
#pragma unroll
      for (int i = 0; i < 8; ++i) {
        float sq = xs[rowb + i] + yc - 2.0f * a.f[i];
        sq = fmaxf(sq, 0.0f);
        out[(size_t)(rowb + i) * NDIM + col] = expf(-sq);
      }
    }
  }
}

extern "C" void kernel_launch(void* const* d_in, const int* in_sizes, int n_in,
                              void* d_out, int out_size, void* d_ws, size_t ws_size,
                              hipStream_t stream) {
  (void)in_sizes; (void)n_in; (void)out_size; (void)ws_size;
  const float* x = (const float*)d_in[0];
  const float* y = (const float*)d_in[1];
  const float* g = (const float*)d_in[2];
  float* out = (float*)d_out;
  float* xs = (float*)d_ws;          // 4096 floats
  float* ys = xs + NDIM;             // 4096 floats  (32 KB total scratch)

  qk_rownorm<<<NDIM / 8, 256, 0, stream>>>(x, g, xs, NDIM);
  qk_rownorm<<<NDIM / 8, 256, 0, stream>>>(y, g, ys, NDIM);

  dim3 grid(NDIM / BN, NDIM / BM);   // 32 x 32 blocks
  qk_gemm<<<grid, 256, 0, stream>>>(x, y, g, xs, ys, out);
}
